// RNN_69114613729165
// MI455X (gfx1250) — matrix operand order repaired and checked by reading, not compiled, not run on hardware
//
#include <hip/hip_runtime.h>
#include <hip/hip_bf16.h>

// Problem sizes (fixed by the reference)
#define RB 64
#define RT 256
#define RI 1024
#define RH 1024
#define RC 1024
#define RM (RB * RT)   // 16384 rows of x / logits

// GEMM tiling
#define BM 128
#define BN 128
#define BK 32
#define LDP 40          // padded LDS row stride in halves (80B, 16B aligned)

typedef __attribute__((ext_vector_type(16))) __bf16 v16bf;
typedef __attribute__((ext_vector_type(8)))  float  v8f;

union Frag16 { int4 q[2]; v16bf v; };

__device__ __forceinline__ unsigned short f32_to_bf16_rne(float f) {
    unsigned int u = __float_as_uint(f);
    unsigned int r = u + 0x7FFFu + ((u >> 16) & 1u);
    return (unsigned short)(r >> 16);
}

// ---------------------------------------------------------------- utilities
__global__ void cast_f32_to_bf16(const float* __restrict__ in,
                                 unsigned short* __restrict__ out, int n4) {
    int i = blockIdx.x * blockDim.x + threadIdx.x;
    if (i >= n4) return;
    float4 f = *(const float4*)(in + (size_t)i * 4);
    ushort4 o;
    o.x = f32_to_bf16_rne(f.x); o.y = f32_to_bf16_rne(f.y);
    o.z = f32_to_bf16_rne(f.z); o.w = f32_to_bf16_rne(f.w);
    *(ushort4*)(out + (size_t)i * 4) = o;
}

__global__ void bias_sum_k(const float* __restrict__ a, const float* __restrict__ b,
                           float* __restrict__ o, int n) {
    int i = blockIdx.x * blockDim.x + threadIdx.x;
    if (i < n) o[i] = a[i] + b[i];
}

__global__ void zero_u16(unsigned short* __restrict__ p, int n) {
    int i = blockIdx.x * blockDim.x + threadIdx.x;
    if (i < n) p[i] = 0;
}

// ---------------------------------------------------- big GEMM: xp = x@Wih^T + bias
__global__ __launch_bounds__(256)
void gemm_xp(const unsigned short* __restrict__ A,   // [RM, RI] bf16 (x)
             const unsigned short* __restrict__ W,   // [RH, RI] bf16 (W_ih, N-major)
             const float* __restrict__ bsum,         // [RH]
             float* __restrict__ Out)                // [RM, RH]
{
    __shared__ unsigned short sA[BM * LDP];
    __shared__ unsigned short sB[BN * LDP];

    const int tid  = threadIdx.x;
    const int lane = tid & 31;
    const int wave = tid >> 5;
    const int half = lane >> 4;
    const int m0 = blockIdx.y * BM;
    const int n0 = blockIdx.x * BN;
    const int wm = wave >> 2;        // 0..1 -> 64 M rows
    const int wn = wave & 3;         // 0..3 -> 32 N cols

    v8f acc[4][2];
    #pragma unroll
    for (int i = 0; i < 4; ++i)
        #pragma unroll
        for (int j = 0; j < 2; ++j) { v8f z = {}; acc[i][j] = z; }

    const int ldr = tid >> 1;              // tile row 0..127
    const int ldc = (tid & 1) * 16;        // half-offset in K (0 or 16)
    const unsigned short* gA = A + (size_t)(m0 + ldr) * RI + ldc;
    const unsigned short* gB = W + (size_t)(n0 + ldr) * RI + ldc;
    unsigned short* lA = &sA[ldr * LDP + ldc];
    unsigned short* lB = &sB[ldr * LDP + ldc];

    for (int k0 = 0; k0 < RI; k0 += BK) {
        if (k0 + BK < RI) {                          // prefetch next K tile (global_prefetch_b8)
            __builtin_prefetch(gA + BK, 0, 1);
            __builtin_prefetch(gB + BK, 0, 1);
        }
        int4 a0 = *(const int4*)(gA);
        int4 a1 = *(const int4*)(gA + 8);
        int4 b0 = *(const int4*)(gB);
        int4 b1 = *(const int4*)(gB + 8);
        gA += BK; gB += BK;
        __syncthreads();
        *(int4*)(lA)     = a0; *(int4*)(lA + 8) = a1;
        *(int4*)(lB)     = b0; *(int4*)(lB + 8) = b1;
        __syncthreads();

        Frag16 af[4];
        #pragma unroll
        for (int tm = 0; tm < 4; ++tm) {
            const unsigned short* p = &sA[(wm * 64 + tm * 16 + (lane & 15)) * LDP + half * 8];
            af[tm].q[0] = *(const int4*)(p);
            af[tm].q[1] = *(const int4*)(p + 16);
        }
        Frag16 bf2[2];
        #pragma unroll
        for (int tn = 0; tn < 2; ++tn) {
            const unsigned short* p = &sB[(wn * 32 + tn * 16 + (lane & 15)) * LDP + half * 8];
            bf2[tn].q[0] = *(const int4*)(p);
            bf2[tn].q[1] = *(const int4*)(p + 16);
        }
        #pragma unroll
        for (int tm = 0; tm < 4; ++tm)
            #pragma unroll
            for (int tn = 0; tn < 2; ++tn)
                acc[tm][tn] = __builtin_amdgcn_wmma_f32_16x16x32_bf16(
                    false, af[tm].v, false, bf2[tn].v,
                    (short)0, acc[tm][tn], false, false);
    }

    #pragma unroll
    for (int tm = 0; tm < 4; ++tm)
        #pragma unroll
        for (int tn = 0; tn < 2; ++tn) {
            const int col  = n0 + wn * 32 + tn * 16 + (lane & 15);
            const int rowb = m0 + wm * 64 + tm * 16 + half * 8;
            const float bv = bsum[col];
            #pragma unroll
            for (int r = 0; r < 8; ++r)
                Out[(size_t)(rowb + r) * RH + col] = acc[tm][tn][r] + bv;
        }
}

// ------------------------------------- recurrent step: h' = tanh(xp_t + h@Whh^T)
__global__ __launch_bounds__(256)
void rnn_step(unsigned short* __restrict__ hs,       // [(T+1), B, H] bf16
              const unsigned short* __restrict__ W,  // [RH, RH] bf16 (W_hh)
              const float* __restrict__ xp,          // [B*T, H] f32
              int t)
{
    const unsigned short* hprev = hs + (size_t)t * RB * RH;
    unsigned short* hnext       = hs + (size_t)(t + 1) * RB * RH;

    __shared__ unsigned short sA[64 * LDP];
    __shared__ unsigned short sB[64 * LDP];

    const int tid  = threadIdx.x;
    const int lane = tid & 31;
    const int wave = tid >> 5;
    const int half = lane >> 4;
    const int n0 = blockIdx.x * 64;
    const int wm = wave & 3;          // 4 M tiles of 16 rows (B=64)
    const int wn = wave >> 2;         // 0..1 -> 32 N cols

    v8f acc[2];
    { v8f z = {}; acc[0] = z; acc[1] = z; }

    const int ldr = tid >> 2;            // 0..63
    const int ldc = (tid & 3) * 8;       // 0,8,16,24
    const unsigned short* gA = hprev + (size_t)ldr * RH + ldc;
    const unsigned short* gB = W + (size_t)(n0 + ldr) * RH + ldc;
    unsigned short* lA = &sA[ldr * LDP + ldc];
    unsigned short* lB = &sB[ldr * LDP + ldc];

    for (int k0 = 0; k0 < RH; k0 += BK) {
        if (k0 + BK < RH) {
            __builtin_prefetch(gA + BK, 0, 1);
            __builtin_prefetch(gB + BK, 0, 1);
        }
        int4 a = *(const int4*)(gA);
        int4 b = *(const int4*)(gB);
        gA += BK; gB += BK;
        __syncthreads();
        *(int4*)(lA) = a;
        *(int4*)(lB) = b;
        __syncthreads();

        Frag16 af;
        {
            const unsigned short* p = &sA[(wm * 16 + (lane & 15)) * LDP + half * 8];
            af.q[0] = *(const int4*)(p);
            af.q[1] = *(const int4*)(p + 16);
        }
        Frag16 bf2[2];
        #pragma unroll
        for (int tn = 0; tn < 2; ++tn) {
            const unsigned short* p = &sB[(wn * 32 + tn * 16 + (lane & 15)) * LDP + half * 8];
            bf2[tn].q[0] = *(const int4*)(p);
            bf2[tn].q[1] = *(const int4*)(p + 16);
        }
        #pragma unroll
        for (int tn = 0; tn < 2; ++tn)
            acc[tn] = __builtin_amdgcn_wmma_f32_16x16x32_bf16(
                false, af.v, false, bf2[tn].v, (short)0, acc[tn], false, false);
    }

    #pragma unroll
    for (int tn = 0; tn < 2; ++tn) {
        const int col = n0 + wn * 32 + tn * 16 + (lane & 15);
        const int rb  = wm * 16 + half * 8;            // batch row
        #pragma unroll
        for (int r = 0; r < 8; ++r) {
            const int brow = rb + r;
            const float pre = xp[(size_t)(brow * RT + t) * RH + col] + acc[tn][r];
            hnext[(size_t)brow * RH + col] = f32_to_bf16_rne(tanhf(pre));
        }
    }
}

// -------------------------------------- logits = hs@Wfc^T + b_fc  (into d_out)
__global__ __launch_bounds__(256)
void gemm_logits(const unsigned short* __restrict__ hs,  // [(T+1), B, H] bf16
                 const unsigned short* __restrict__ W,   // [RC, RH] bf16 (W_fc)
                 const float* __restrict__ bfc,          // [RC]
                 float* __restrict__ Out)                // [RM, RC]
{
    __shared__ unsigned short sA[BM * LDP];
    __shared__ unsigned short sB[BN * LDP];

    const int tid  = threadIdx.x;
    const int lane = tid & 31;
    const int wave = tid >> 5;
    const int half = lane >> 4;
    const int m0 = blockIdx.y * BM;
    const int n0 = blockIdx.x * BN;
    const int wm = wave >> 2;
    const int wn = wave & 3;

    v8f acc[4][2];
    #pragma unroll
    for (int i = 0; i < 4; ++i)
        #pragma unroll
        for (int j = 0; j < 2; ++j) { v8f z = {}; acc[i][j] = z; }

    const int ldr = tid >> 1;
    const int ldc = (tid & 1) * 16;
    // Row m = b*T + t maps to hs slice (t+1), batch b
    const int m = m0 + ldr;
    const unsigned short* gA =
        hs + ((size_t)((m & (RT - 1)) + 1) * RB + (m >> 8)) * RH + ldc;
    const unsigned short* gB = W + (size_t)(n0 + ldr) * RH + ldc;
    unsigned short* lA = &sA[ldr * LDP + ldc];
    unsigned short* lB = &sB[ldr * LDP + ldc];

    for (int k0 = 0; k0 < RH; k0 += BK) {
        if (k0 + BK < RH) {
            __builtin_prefetch(gA + BK, 0, 1);
            __builtin_prefetch(gB + BK, 0, 1);
        }
        int4 a0 = *(const int4*)(gA);
        int4 a1 = *(const int4*)(gA + 8);
        int4 b0 = *(const int4*)(gB);
        int4 b1 = *(const int4*)(gB + 8);
        gA += BK; gB += BK;
        __syncthreads();
        *(int4*)(lA)     = a0; *(int4*)(lA + 8) = a1;
        *(int4*)(lB)     = b0; *(int4*)(lB + 8) = b1;
        __syncthreads();

        Frag16 af[4];
        #pragma unroll
        for (int tm = 0; tm < 4; ++tm) {
            const unsigned short* p = &sA[(wm * 64 + tm * 16 + (lane & 15)) * LDP + half * 8];
            af[tm].q[0] = *(const int4*)(p);
            af[tm].q[1] = *(const int4*)(p + 16);
        }
        Frag16 bf2[2];
        #pragma unroll
        for (int tn = 0; tn < 2; ++tn) {
            const unsigned short* p = &sB[(wn * 32 + tn * 16 + (lane & 15)) * LDP + half * 8];
            bf2[tn].q[0] = *(const int4*)(p);
            bf2[tn].q[1] = *(const int4*)(p + 16);
        }
        #pragma unroll
        for (int tm = 0; tm < 4; ++tm)
            #pragma unroll
            for (int tn = 0; tn < 2; ++tn)
                acc[tm][tn] = __builtin_amdgcn_wmma_f32_16x16x32_bf16(
                    false, af[tm].v, false, bf2[tn].v,
                    (short)0, acc[tm][tn], false, false);
    }

    #pragma unroll
    for (int tm = 0; tm < 4; ++tm)
        #pragma unroll
        for (int tn = 0; tn < 2; ++tn) {
            const int col  = n0 + wn * 32 + tn * 16 + (lane & 15);
            const int rowb = m0 + wm * 64 + tm * 16 + half * 8;
            const float bv = bfc[col];
            #pragma unroll
            for (int r = 0; r < 8; ++r)
                Out[(size_t)(rowb + r) * RC + col] = acc[tm][tn][r] + bv;
        }
}

// ------------------------------------------------ in-place row log-softmax
__global__ __launch_bounds__(256)
void log_softmax_rows(float* __restrict__ logits) {
    float* p = logits + (size_t)blockIdx.x * RC;
    const int tid  = threadIdx.x;
    const int lane = tid & 31;
    const int wave = tid >> 5;

    float4 v = *(const float4*)(p + tid * 4);
    float m = fmaxf(fmaxf(v.x, v.y), fmaxf(v.z, v.w));
    #pragma unroll
    for (int off = 16; off > 0; off >>= 1) m = fmaxf(m, __shfl_xor(m, off, 32));

    __shared__ float red[8];
    if (lane == 0) red[wave] = m;
    __syncthreads();
    float bm = red[0];
    #pragma unroll
    for (int i = 1; i < 8; ++i) bm = fmaxf(bm, red[i]);

    float s = expf(v.x - bm) + expf(v.y - bm) + expf(v.z - bm) + expf(v.w - bm);
    #pragma unroll
    for (int off = 16; off > 0; off >>= 1) s += __shfl_xor(s, off, 32);
    __syncthreads();
    if (lane == 0) red[wave] = s;
    __syncthreads();
    float bs = 0.f;
    #pragma unroll
    for (int i = 0; i < 8; ++i) bs += red[i];

    const float lse = bm + logf(bs);
    v.x -= lse; v.y -= lse; v.z -= lse; v.w -= lse;
    *(float4*)(p + tid * 4) = v;
}

// ---------------------------------------------------------------- launcher
extern "C" void kernel_launch(void* const* d_in, const int* in_sizes, int n_in,
                              void* d_out, int out_size, void* d_ws, size_t ws_size,
                              hipStream_t stream) {
    const float* x   = (const float*)d_in[0];
    const float* Wih = (const float*)d_in[1];
    const float* Whh = (const float*)d_in[2];
    const float* bih = (const float*)d_in[3];
    const float* bhh = (const float*)d_in[4];
    const float* Wfc = (const float*)d_in[5];
    const float* bfc = (const float*)d_in[6];
    float* out = (float*)d_out;

    char* ws = (char*)d_ws;
    size_t off = 0;
    auto alloc = [&](size_t bytes) -> void* {
        void* p = ws + off;
        off += (bytes + 255) & ~(size_t)255;
        return p;
    };
    unsigned short* x_bf   = (unsigned short*)alloc((size_t)RM * RI * 2);   // 32 MB
    unsigned short* Wih_bf = (unsigned short*)alloc((size_t)RH * RI * 2);   //  2 MB
    unsigned short* Whh_bf = (unsigned short*)alloc((size_t)RH * RH * 2);   //  2 MB
    unsigned short* Wfc_bf = (unsigned short*)alloc((size_t)RC * RH * 2);   //  2 MB
    float*          bsum   = (float*)alloc((size_t)RH * 4);                 //  4 KB
    float*          xp     = (float*)alloc((size_t)RM * RH * 4);            // 64 MB
    unsigned short* hs     = (unsigned short*)alloc((size_t)(RT + 1) * RB * RH * 2); // ~32 MB
    (void)ws_size; (void)in_sizes; (void)n_in; (void)out_size;

    // Precision casts (one pass each)
    cast_f32_to_bf16<<<(RM * RI / 4 + 255) / 256, 256, 0, stream>>>(x, x_bf, RM * RI / 4);
    cast_f32_to_bf16<<<(RH * RI / 4 + 255) / 256, 256, 0, stream>>>(Wih, Wih_bf, RH * RI / 4);
    cast_f32_to_bf16<<<(RH * RH / 4 + 255) / 256, 256, 0, stream>>>(Whh, Whh_bf, RH * RH / 4);
    cast_f32_to_bf16<<<(RC * RH / 4 + 255) / 256, 256, 0, stream>>>(Wfc, Wfc_bf, RC * RH / 4);
    bias_sum_k<<<(RH + 255) / 256, 256, 0, stream>>>(bih, bhh, bsum, RH);
    zero_u16<<<(RB * RH + 255) / 256, 256, 0, stream>>>(hs, RB * RH);      // h0 = 0

    // xp = x @ W_ih^T + (b_ih + b_hh)
    gemm_xp<<<dim3(RH / BN, RM / BM), 256, 0, stream>>>(x_bf, Wih_bf, bsum, xp);

    // Serial scan: 256 dependent WMMA steps, W_hh resident in L2
    for (int t = 0; t < RT; ++t)
        rnn_step<<<dim3(RH / 64), 256, 0, stream>>>(hs, Whh_bf, xp, t);

    // logits into d_out, then in-place log-softmax
    gemm_logits<<<dim3(RC / BN, RM / BM), 256, 0, stream>>>(hs, Wfc_bf, bfc, out);
    log_softmax_rows<<<RM, 256, 0, stream>>>(out);
}
